// TexualSelectedEmbeddingLayer_21036749815975
// MI455X (gfx1250) — matrix operand order repaired and checked by reading, not compiled, not run on hardware
//
#include <hip/hip_runtime.h>
#include <hip/hip_bf16.h>
#include <math.h>

// ---------------------------------------------------------------------------
// Problem constants (from the reference)
// ---------------------------------------------------------------------------
#define B    128
#define L    512
#define D    512
#define E    1024
#define H    512
#define K    255
#define KP   256                 // K padded to 256 rows per batch (row 255 = zero pad)
#define ROWS (B * KP)            // 32768 flattened rows (32640 valid)
#define NVALID ((float)(B * K))  // 32640 rows feed the BatchNorm statistics

typedef __attribute__((ext_vector_type(16))) __bf16 v16bf;
typedef __attribute__((ext_vector_type(8)))  float  v8f;

union Frag { v16bf v; uint4 q[2]; };

__device__ __forceinline__ unsigned short f32_to_bf16(float f) {
    union { float f; unsigned u; } x; x.f = f;
    unsigned r = x.u + 0x7FFFu + ((x.u >> 16) & 1u);   // round-to-nearest-even
    return (unsigned short)(r >> 16);
}
__device__ __forceinline__ float bf16_to_f32(unsigned short s) {
    union { unsigned u; float f; } x; x.u = ((unsigned)s) << 16;
    return x.f;
}

// A-fragment: activations [M,K] row-major. Lane layout per ISA 16-bit A 16x32:
//   lane = hi*16+m ; elems 0..7 -> K = 8*hi+e ; elems 8..15 -> K = 16+8*hi+(e-8)
__device__ __forceinline__ v16bf load_a_frag(const unsigned short* row, int k0, int hi) {
    Frag f;
    f.q[0] = *(const uint4*)(row + k0 +      8 * hi);
    f.q[1] = *(const uint4*)(row + k0 + 16 + 8 * hi);
    return f.v;
}
// B-fragment: weights stored [N,K] row-major (B[k,n] = W[n,k]). Per ISA 16-bit B:
//   lane = hi*16+n ; elems 0..7 -> K = 8*hi+e ; elems 8..15 -> K = 16+8*hi+(e-8)
//   -> two 16-byte chunks of weight row n at K offsets 8*hi and 16+8*hi.
__device__ __forceinline__ v16bf load_b_frag(const unsigned short* row, int k0, int hi) {
    Frag f;
    f.q[0] = *(const uint4*)(row + k0 +      8 * hi);
    f.q[1] = *(const uint4*)(row + k0 + 16 + 8 * hi);
    return f.v;
}

#define WMMA_BF16(A, Bf, C) \
    __builtin_amdgcn_wmma_f32_16x16x32_bf16(false, (A), false, (Bf), (short)0, (C), false, false)

// ---------------------------------------------------------------------------
// K0: convert the three weight matrices to bf16 (w1: HxD, fc_w: ExD, w2: ExH)
// ---------------------------------------------------------------------------
__global__ void prep_weights(const float* w1, const float* fcw, const float* w2,
                             unsigned short* w1b, unsigned short* fcwb, unsigned short* w2b) {
    const int N1 = H * D, N2 = E * D, N3 = E * H;
    int stride = gridDim.x * blockDim.x;
    for (int g = blockIdx.x * blockDim.x + threadIdx.x; g < N1 + N2 + N3; g += stride) {
        if (g < N1)           w1b[g]            = f32_to_bf16(w1[g]);
        else if (g < N1 + N2) fcwb[g - N1]      = f32_to_bf16(fcw[g - N1]);
        else                  w2b[g - N1 - N2]  = f32_to_bf16(w2[g - N1 - N2]);
    }
}

// ---------------------------------------------------------------------------
// K1: per-batch mask/lengths/eos + masked attention row + exact top-K ranking.
// One block of 512 threads per batch; rank = #elements strictly greater
// (index as tiebreak) reproduces jax.lax.top_k's sorted order.
// ---------------------------------------------------------------------------
__global__ void select_topk(const int* __restrict__ text, const float* __restrict__ att,
                            int* __restrict__ idx, int* __restrict__ lens) {
    const int b = blockIdx.x, t = threadIdx.x;
    __shared__ int   sv[L], si[L], ss[L];
    __shared__ float srow[L];

    int tok = text[b * L + t];
    int mk  = (tok != 0) ? 1 : 0;
    sv[t] = tok; si[t] = t; ss[t] = mk;
    __syncthreads();
    for (int s = L / 2; s > 0; s >>= 1) {             // argmax (first-max) + mask sum
        if (t < s) {
            ss[t] += ss[t + s];
            int v2 = sv[t + s], i2 = si[t + s];
            if (v2 > sv[t] || (v2 == sv[t] && i2 < si[t])) { sv[t] = v2; si[t] = i2; }
        }
        __syncthreads();
    }
    const int eos = si[0];
    const int nmask = ss[0];

    float v = att[((size_t)b * L + eos) * L + t];
    if (t == 0 || t == eos) v = -1.0f;                 // kill col 0 and the EOS col
    v *= (float)mk;
    srow[t] = v;
    __syncthreads();

    int cnt = 0;
    for (int j = 0; j < L; ++j) {
        float x = srow[j];
        cnt += (x > v || (x == v && j < t)) ? 1 : 0;
    }
    if (cnt < K) idx[b * KP + cnt] = t;
    if (t == 0)  lens[b] = min(nmask - 2, K);
}

// ---------------------------------------------------------------------------
// K2: gather selected token features + L2 normalize, store bf16 [B,256,D].
// Row 255 of every batch is the zero pad row.
// ---------------------------------------------------------------------------
__global__ void gather_norm(const float* __restrict__ tf, const int* __restrict__ idx,
                            unsigned short* __restrict__ base) {
    const int blk = blockIdx.x;          // b*256 + k
    const int b = blk >> 8, k = blk & 255, t = threadIdx.x;
    unsigned short* dst = base + (size_t)blk * D;
    if (k == K) { dst[t] = 0; dst[t + 256] = 0; return; }   // pad row -> zeros

    const int sel = idx[b * KP + k];
    const float* src = tf + ((size_t)b * L + sel) * D;
    float x0 = src[t], x1 = src[t + 256];

    __shared__ float red[256];
    red[t] = x0 * x0 + x1 * x1;
    __syncthreads();
    for (int s = 128; s > 0; s >>= 1) { if (t < s) red[t] += red[t + s]; __syncthreads(); }
    const float inv = 1.0f / (sqrtf(red[0]) + 1e-8f);
    dst[t]       = f32_to_bf16(x0 * inv);
    dst[t + 256] = f32_to_bf16(x1 * inv);
}

// ---------------------------------------------------------------------------
// K3: GEMM1  h = base @ w1^T + b1   (32768x512 @ 512x512), bf16 WMMA, f32 acc.
// 32x32 register-blocked tile per wave (2 M-frags x 2 N-frags, 4 accumulators):
// 8 global_load_b128 feed 4 v_wmma per K-step -> 2 loads/WMMA (vs 4 unblocked).
// ---------------------------------------------------------------------------
__global__ void gemm1(const unsigned short* __restrict__ base,
                      const unsigned short* __restrict__ w1b,
                      const float* __restrict__ b1,
                      unsigned short* __restrict__ h) {
    const int wid  = blockIdx.x * 4 + (threadIdx.x >> 5);
    const int lane = threadIdx.x & 31;
    const int mBlk = wid >> 4, nBlk = wid & 15;        // 1024 x 16 blocks of 32x32
    const int row0 = mBlk * 32, n0 = nBlk * 32;
    const int m = lane & 15, hi = lane >> 4;

    const unsigned short* arow0 = base + (size_t)(row0 + m) * D;
    const unsigned short* arow1 = arow0 + (size_t)16 * D;
    const unsigned short* brow0 = w1b + (size_t)(n0 + m) * D;
    const unsigned short* brow1 = brow0 + (size_t)16 * D;

    v8f c00 = {}, c01 = {}, c10 = {}, c11 = {};
    for (int k0 = 0; k0 < D; k0 += 32) {
        v16bf a0 = load_a_frag(arow0, k0, hi);
        v16bf a1 = load_a_frag(arow1, k0, hi);
        v16bf b0 = load_b_frag(brow0, k0, hi);
        v16bf b1f = load_b_frag(brow1, k0, hi);
        c00 = WMMA_BF16(a0, b0, c00);
        c01 = WMMA_BF16(a0, b1f, c01);
        c10 = WMMA_BF16(a1, b0, c10);
        c11 = WMMA_BF16(a1, b1f, c11);
    }
    const float bias0 = b1[n0 + m], bias1 = b1[n0 + 16 + m];
    for (int r = 0; r < 8; ++r) {
        int rowA = row0 + r + 8 * hi;                  // C/D layout: M = r + 8*hi, N = lane&15
        int rowB = rowA + 16;
        h[(size_t)rowA * H + (n0 + m)]      = f32_to_bf16(c00[r] + bias0);
        h[(size_t)rowA * H + (n0 + 16 + m)] = f32_to_bf16(c01[r] + bias1);
        h[(size_t)rowB * H + (n0 + m)]      = f32_to_bf16(c10[r] + bias0);
        h[(size_t)rowB * H + (n0 + 16 + m)] = f32_to_bf16(c11[r] + bias1);
    }
}

// ---------------------------------------------------------------------------
// K4: BatchNorm statistics over the 32640 valid rows (deterministic: one block
// per column, fixed-order tree reduction; h lives in L2 so strided reads hit).
// ---------------------------------------------------------------------------
__global__ void bn_stats(const unsigned short* __restrict__ h,
                         float* __restrict__ mu, float* __restrict__ var) {
    const int col = blockIdx.x, t = threadIdx.x;
    float s = 0.0f, s2 = 0.0f;
    for (int row = t; row < ROWS; row += 256) {
        if ((row & 255) == K) continue;                // skip pad rows
        float x = bf16_to_f32(h[(size_t)row * H + col]);
        s += x; s2 += x * x;
    }
    __shared__ float rs[256], rs2[256];
    rs[t] = s; rs2[t] = s2;
    __syncthreads();
    for (int st = 128; st > 0; st >>= 1) {
        if (t < st) { rs[t] += rs[t + st]; rs2[t] += rs2[t + st]; }
        __syncthreads();
    }
    if (t == 0) {
        float mean = rs[0] / NVALID;
        mu[col]  = mean;
        var[col] = rs2[0] / NVALID - mean * mean;      // biased variance (torch BN)
    }
}

// ---------------------------------------------------------------------------
// K5: normalize + affine + ReLU, in place on the bf16 h buffer.
// ---------------------------------------------------------------------------
__global__ void bn_relu(unsigned short* __restrict__ h,
                        const float* __restrict__ mu, const float* __restrict__ var,
                        const float* __restrict__ g,  const float* __restrict__ bb) {
    const size_t total = (size_t)ROWS * H;
    const size_t stride = (size_t)gridDim.x * blockDim.x;
    for (size_t i = (size_t)blockIdx.x * blockDim.x + threadIdx.x; i < total; i += stride) {
        int col = (int)(i & (H - 1));
        float x = bf16_to_f32(h[i]);
        float y = (x - mu[col]) * rsqrtf(var[col] + 1e-5f) * g[col] + bb[col];
        h[i] = f32_to_bf16(fmaxf(y, 0.0f));
    }
}

// ---------------------------------------------------------------------------
// K6: fused GEMM2 + fc + biases + masked max pool, 32x32 register-blocked.
//   local = hn @ w2^T + base @ fc_w^T + (b2 + fc_b)
//   pooled[b,n] = max over k < lens[b] of local[b,k,n]
// One wave per (batch, 32-col block): 8 M-blocks of 32 rows; per K-step,
// 8 A-frag/B-frag loads (16 b128) feed 8 WMMAs -> 2 loads/WMMA.
// ---------------------------------------------------------------------------
__global__ void gemm2_fc_pool(const unsigned short* __restrict__ hn,
                              const unsigned short* __restrict__ base,
                              const unsigned short* __restrict__ w2b,
                              const unsigned short* __restrict__ fcwb,
                              const float* __restrict__ b2, const float* __restrict__ fcb,
                              const int* __restrict__ lens, float* __restrict__ out) {
    const int wid  = blockIdx.x * 4 + (threadIdx.x >> 5);
    const int lane = threadIdx.x & 31;
    const int b = wid >> 5, nBlk = wid & 31;           // 128 batches x 32 col-blocks
    const int n0 = nBlk * 32;
    const int m = lane & 15, hi = lane >> 4;

    const int   lensb = lens[b];
    const float bias0 = b2[n0 + m]      + fcb[n0 + m];
    const float bias1 = b2[n0 + 16 + m] + fcb[n0 + 16 + m];
    const unsigned short* wrow0 = w2b  + (size_t)(n0 + m) * H;
    const unsigned short* wrow1 = wrow0 + (size_t)16 * H;
    const unsigned short* frow0 = fcwb + (size_t)(n0 + m) * D;
    const unsigned short* frow1 = frow0 + (size_t)16 * D;

    float lmax0 = -INFINITY, lmax1 = -INFINITY;
    for (int mBlk = 0; mBlk < 8; ++mBlk) {
        const int row0 = b * KP + mBlk * 32;
        const unsigned short* hrow0 = hn + (size_t)(row0 + m) * H;
        const unsigned short* hrow1 = hrow0 + (size_t)16 * H;
        const unsigned short* xrow0 = base + (size_t)(row0 + m) * D;
        const unsigned short* xrow1 = xrow0 + (size_t)16 * D;

        v8f c00 = {}, c01 = {}, c10 = {}, c11 = {};
        for (int k0 = 0; k0 < 512; k0 += 32) {
            v16bf ah0 = load_a_frag(hrow0, k0, hi);
            v16bf ah1 = load_a_frag(hrow1, k0, hi);
            v16bf bw0 = load_b_frag(wrow0, k0, hi);
            v16bf bw1 = load_b_frag(wrow1, k0, hi);
            c00 = WMMA_BF16(ah0, bw0, c00);
            c01 = WMMA_BF16(ah0, bw1, c01);
            c10 = WMMA_BF16(ah1, bw0, c10);
            c11 = WMMA_BF16(ah1, bw1, c11);
            v16bf ax0 = load_a_frag(xrow0, k0, hi);
            v16bf ax1 = load_a_frag(xrow1, k0, hi);
            v16bf bf0 = load_b_frag(frow0, k0, hi);
            v16bf bf1 = load_b_frag(frow1, k0, hi);
            c00 = WMMA_BF16(ax0, bf0, c00);
            c01 = WMMA_BF16(ax0, bf1, c01);
            c10 = WMMA_BF16(ax1, bf0, c10);
            c11 = WMMA_BF16(ax1, bf1, c11);
        }
        for (int r = 0; r < 8; ++r) {
            int kA = mBlk * 32 + r + 8 * hi;           // row index within batch
            int kB = kA + 16;
            if (kA < lensb) {
                lmax0 = fmaxf(lmax0, c00[r] + bias0);
                lmax1 = fmaxf(lmax1, c01[r] + bias1);
            }
            if (kB < lensb) {
                lmax0 = fmaxf(lmax0, c10[r] + bias0);
                lmax1 = fmaxf(lmax1, c11[r] + bias1);
            }
        }
    }
    // lanes L and L^16 hold the same output columns (different rows) -> combine
    lmax0 = fmaxf(lmax0, __shfl_xor(lmax0, 16));
    lmax1 = fmaxf(lmax1, __shfl_xor(lmax1, 16));
    if (hi == 0) {
        out[(size_t)b * E + n0 + m]      = lmax0;
        out[(size_t)b * E + n0 + 16 + m] = lmax1;
    }
}

// ---------------------------------------------------------------------------
// Host-side launch: workspace carve-out + 7-kernel pipeline on `stream`.
// ---------------------------------------------------------------------------
extern "C" void kernel_launch(void* const* d_in, const int* in_sizes, int n_in,
                              void* d_out, int out_size, void* d_ws, size_t ws_size,
                              hipStream_t stream) {
    (void)in_sizes; (void)n_in; (void)out_size; (void)ws_size;
    const float* text_feats = (const float*)d_in[0];
    const int*   text       = (const int*)  d_in[1];
    const float* attenscore = (const float*)d_in[2];
    const float* fc_w       = (const float*)d_in[3];
    const float* fc_b       = (const float*)d_in[4];
    const float* w1         = (const float*)d_in[5];
    const float* b1         = (const float*)d_in[6];
    const float* bn_g       = (const float*)d_in[7];
    const float* bn_b       = (const float*)d_in[8];
    const float* w2         = (const float*)d_in[9];
    const float* b2         = (const float*)d_in[10];
    float*       out        = (float*)d_out;

    char* ws = (char*)d_ws;
    size_t o = 0;
    auto carve = [&](size_t bytes) { char* p = ws + o; o = (o + bytes + 255) & ~(size_t)255; return p; };
    int*            lens = (int*)           carve(B * sizeof(int));
    int*            idx  = (int*)           carve(B * KP * sizeof(int));
    float*          mu   = (float*)         carve(H * sizeof(float));
    float*          var  = (float*)         carve(H * sizeof(float));
    unsigned short* base = (unsigned short*)carve((size_t)ROWS * D * 2);   // 32 MB
    unsigned short* w1b  = (unsigned short*)carve((size_t)H * D * 2);
    unsigned short* fcwb = (unsigned short*)carve((size_t)E * D * 2);
    unsigned short* w2b  = (unsigned short*)carve((size_t)E * H * 2);
    unsigned short* h    = (unsigned short*)carve((size_t)ROWS * H * 2);   // 32 MB

    prep_weights  <<<512, 256, 0, stream>>>(w1, fc_w, w2, w1b, fcwb, w2b);
    select_topk   <<<B, L, 0, stream>>>(text, attenscore, idx, lens);
    gather_norm   <<<ROWS, 256, 0, stream>>>(text_feats, idx, base);
    gemm1         <<<(ROWS / 32) * (H / 32) / 4, 128, 0, stream>>>(base, w1b, b1, h);
    bn_stats      <<<H, 256, 0, stream>>>(h, mu, var);
    bn_relu       <<<4096, 256, 0, stream>>>(h, mu, var, bn_g, bn_b);
    gemm2_fc_pool <<<B * (E / 32) / 4, 128, 0, stream>>>(h, base, w2b, fcwb,
                                                         b2, fc_b, lens, out);
}
